// CLinear_687194767979
// MI455X (gfx1250) — matrix operand order repaired and checked by reading, not compiled, not run on hardware
//
#include <hip/hip_runtime.h>
#include <stdint.h>

// ---- types ----
typedef __attribute__((ext_vector_type(16))) _Float16 v16h;
typedef __attribute__((ext_vector_type(8)))  _Float16 v8h;
typedef __attribute__((ext_vector_type(4)))  _Float16 v4h;
typedef __attribute__((ext_vector_type(8)))  float    v8f;
typedef __attribute__((ext_vector_type(4)))  float    v4f;

// ---- problem dims ----
#define TOKENS_N 4096
#define OFEAT    11008
#define IFEAT    4096
#define GS_      64
#define NGRP     (IFEAT / GS_)   // 64 groups along K
#define PKB      (GS_ / 2)       // 32 packed bytes per group per row

// ---- tiling ----
#define MT   128
#define NT   128
#define LDA  72                  // 64 + 8 pad (halves): 144B rows, 16B aligned
#define LDB  72

// Convert staged registers into LDS tiles (f16) for one K-group.
__device__ __forceinline__ void stage_store(
    _Float16 (*__restrict__ Asb)[LDA],
    _Float16 (*__restrict__ Bsb)[LDB],
    const v4f xr[8], uint4 pw, float m0, float s0,
    int xrow0, int xcol4, int wr, int wseg)
{
    #pragma unroll
    for (int i = 0; i < 8; ++i) {
        v4h h;
        h.x = (_Float16)xr[i].x; h.y = (_Float16)xr[i].y;
        h.z = (_Float16)xr[i].z; h.w = (_Float16)xr[i].w;
        *(v4h*)&Asb[xrow0 + i * 16][xcol4] = h;
    }
    union { uint4 v; uint8_t b[16]; } u;
    u.v = pw;
    v8h hi0, hi1, lo0, lo1;
    #pragma unroll
    for (int i = 0; i < 8; ++i) {
        const uint8_t b0 = u.b[i];
        const uint8_t b1 = u.b[i + 8];
        hi0[i] = (_Float16)((float)(b0 >> 4)  * s0 + m0);
        lo0[i] = (_Float16)((float)(b0 & 15) * s0 + m0);
        hi1[i] = (_Float16)((float)(b1 >> 4)  * s0 + m0);
        lo1[i] = (_Float16)((float)(b1 & 15) * s0 + m0);
    }
    // hi nibble -> group positions [0,32), lo nibble -> [32,64)
    *(v8h*)&Bsb[wr][wseg]      = hi0;
    *(v8h*)&Bsb[wr][wseg + 8]  = hi1;
    *(v8h*)&Bsb[wr][wseg + 32] = lo0;
    *(v8h*)&Bsb[wr][wseg + 40] = lo1;
}

__global__ __launch_bounds__(256) void qgemm_wmma_f16(
    const float*   __restrict__ x,        // [TOKENS, IFEAT] f32
    const uint8_t* __restrict__ packed,   // [OFEAT, NGRP, PKB] u8
    const float*   __restrict__ mn,       // [OFEAT, NGRP] f32
    const float*   __restrict__ inv_scale,// [OFEAT, NGRP] f32
    const float*   __restrict__ bias,     // [OFEAT] f32
    float*         __restrict__ out)      // [TOKENS, OFEAT] f32
{
    __shared__ _Float16 As[2][MT][LDA];   // double-buffered activations
    __shared__ _Float16 Bs[2][NT][LDB];   // double-buffered dequant weights

    const int nb   = blockIdx.x;          // 0..85  (N blocks)
    const int mb   = blockIdx.y;          // 0..31  (M blocks)
    const int tid  = threadIdx.x;
    const int lane = tid & 31;
    const int wave = tid >> 5;            // 0..7
    const int mwave = (wave & 3) * 32;
    const int nwave = (wave >> 2) * 64;
    const int lr = lane & 15;
    const int lh = lane >> 4;

    // X staging geometry: 128 rows x 64 f32 cols, 256 threads x 8 float4
    const int xcol4 = (tid & 15) * 4;
    const int xrow0 = tid >> 4;

    // W staging geometry: 128 rows x 32 bytes; each thread does 16 bytes
    const int wr   = tid >> 1;
    const int wseg = (tid & 1) * 16;
    const int orow = nb * NT + wr;

    const float* xbase = x + (size_t)(mb * MT) * IFEAT;

    v8f acc[2][4];
    {
        v8f z = {};
        #pragma unroll
        for (int mt = 0; mt < 2; ++mt)
            #pragma unroll
            for (int nt = 0; nt < 4; ++nt)
                acc[mt][nt] = z;
    }

    // staging registers
    v4f  xr[8];
    uint4 pw;
    float m0, s0;

    // ---- prologue: load + stage group 0 ----
    #pragma unroll
    for (int i = 0; i < 8; ++i)
        xr[i] = *(const v4f*)(xbase + (size_t)(xrow0 + i * 16) * IFEAT + xcol4);
    pw = *(const uint4*)(packed + (size_t)orow * NGRP * PKB + wseg);
    m0 = mn[(size_t)orow * NGRP];
    s0 = inv_scale[(size_t)orow * NGRP];
    stage_store(As[0], Bs[0], xr, pw, m0, s0, xrow0, xcol4, wr, wseg);
    __syncthreads();

    for (int g = 0; g < NGRP; ++g) {
        const int cur = g & 1;

        // ---- issue next group's global loads (overlap with WMMAs) ----
        if (g + 1 < NGRP) {
            const float* xsrc = xbase + (size_t)(g + 1) * GS_;
            #pragma unroll
            for (int i = 0; i < 8; ++i)
                xr[i] = *(const v4f*)(xsrc + (size_t)(xrow0 + i * 16) * IFEAT + xcol4);
            pw = *(const uint4*)(packed + ((size_t)orow * NGRP + g + 1) * PKB + wseg);
            m0 = mn[(size_t)orow * NGRP + g + 1];
            s0 = inv_scale[(size_t)orow * NGRP + g + 1];
            if (g + 2 < NGRP) {
                __builtin_prefetch(xsrc + GS_ + (size_t)xrow0 * IFEAT + xcol4, 0, 1);
                __builtin_prefetch(packed + ((size_t)orow * NGRP + g + 2) * PKB + wseg, 0, 1);
            }
        }

        // ---- WMMA on current buffer: two k-steps of 32 ----
        #pragma unroll
        for (int ks = 0; ks < 2; ++ks) {
            const int k0 = ks * 32;
            v16h af[2];
            #pragma unroll
            for (int mt = 0; mt < 2; ++mt) {
                const int m = mwave + mt * 16 + lr;
                // A layout: lanes 0-15 K=0..7/16..23, lanes 16-31 K=8..15/24..31
                v8h a0 = *(const v8h*)&As[cur][m][k0 + lh * 8];
                v8h a1 = *(const v8h*)&As[cur][m][k0 + 16 + lh * 8];
                #pragma unroll
                for (int i = 0; i < 8; ++i) { af[mt][i] = a0[i]; af[mt][i + 8] = a1[i]; }
            }
            v16h bf[4];
            #pragma unroll
            for (int nt = 0; nt < 4; ++nt) {
                const int n = nwave + nt * 16 + lr;
                // B layout: lanes 0-15 K=0..7/16..23, lanes 16-31 K=8..15/24..31
                v8h b0 = *(const v8h*)&Bs[cur][n][k0 + lh * 8];
                v8h b1 = *(const v8h*)&Bs[cur][n][k0 + 16 + lh * 8];
                #pragma unroll
                for (int i = 0; i < 8; ++i) { bf[nt][i] = b0[i]; bf[nt][i + 8] = b1[i]; }
            }
            #pragma unroll
            for (int mt = 0; mt < 2; ++mt)
                #pragma unroll
                for (int nt = 0; nt < 4; ++nt)
                    acc[mt][nt] = __builtin_amdgcn_wmma_f32_16x16x32_f16(
                        false, af[mt], false, bf[nt],
                        (short)0, acc[mt][nt], false, false);
        }

        // ---- stage next group into the other buffer ----
        if (g + 1 < NGRP) {
            const int nxt = cur ^ 1;
            stage_store(As[nxt], Bs[nxt], xr, pw, m0, s0, xrow0, xcol4, wr, wseg);
        }
        __syncthreads();
    }

    // ---- epilogue: C layout lanes0-15 M=v, lanes16-31 M=8+v ----
    const int row0 = mb * MT + mwave + lh * 8;
    #pragma unroll
    for (int nt = 0; nt < 4; ++nt) {
        const int col = nb * NT + nwave + nt * 16 + lr;
        const float bv = bias[col];
        #pragma unroll
        for (int mt = 0; mt < 2; ++mt) {
            #pragma unroll
            for (int v = 0; v < 8; ++v) {
                const int row = row0 + mt * 16 + v;
                out[(size_t)row * OFEAT + col] = acc[mt][nt][v] + bv;
            }
        }
    }
}

extern "C" void kernel_launch(void* const* d_in, const int* in_sizes, int n_in,
                              void* d_out, int out_size, void* d_ws, size_t ws_size,
                              hipStream_t stream) {
    const float*   x    = (const float*)d_in[0];
    const uint8_t* pk   = (const uint8_t*)d_in[1];
    const float*   mn   = (const float*)d_in[2];
    const float*   isc  = (const float*)d_in[3];
    const float*   bias = (const float*)d_in[4];
    float* out = (float*)d_out;

    dim3 grid(OFEAT / NT, TOKENS_N / MT);  // (86, 32)
    qgemm_wmma_f16<<<grid, dim3(256), 0, stream>>>(x, pk, mn, isc, bias, out);
}
